// VanillaRNN_42640435314711
// MI455X (gfx1250) — matrix operand order repaired and checked by reading, not compiled, not run on hardware
//
#include <hip/hip_runtime.h>

#define SEQ      128
#define DIM      64
#define HID      256
#define NCLS     10
#define BATCH    4096
#define BM       32
#define THREADS  512
#define WHH_S    (HID + 8)   // 264 halves -> 528B row stride, conflict-free b128 reads
#define XS_S     (DIM + 8)   // 72 halves
#define HS_S     (HID + 8)   // 264 halves

typedef _Float16 half4  __attribute__((ext_vector_type(4)));
typedef _Float16 half8  __attribute__((ext_vector_type(8)));
typedef _Float16 half16 __attribute__((ext_vector_type(16)));
typedef float    v8f    __attribute__((ext_vector_type(8)));

union FragU { half16 v; half8 h[2]; _Float16 e[16]; };

// A fragment (16x32 f16): lane<16 holds K = kb..kb+15,
// lane>=16 holds K = kb+16..kb+31 (contiguous, matches B).
__device__ __forceinline__ half16 frag_lds_a(const _Float16* row, int kb, int hi) {
  FragU u;
  u.h[0] = *(const half8*)(row + kb + hi * 16);
  u.h[1] = *(const half8*)(row + kb + hi * 16 + 8);
  return u.v;
}
// B fragment (32x16 f16), B = W^T => lane holds 16 contiguous K of W's row n:
// lane<16: K = kb..kb+15, lane>=16: K = kb+16..kb+31.
__device__ __forceinline__ half16 frag_lds_b(const _Float16* row, int kb, int hi) {
  const _Float16* p = row + kb + hi * 16;
  FragU u;
  u.h[0] = *(const half8*)(p);
  u.h[1] = *(const half8*)(p + 8);
  return u.v;
}
__device__ __forceinline__ half16 frag_gl_b_f32(const float* p) {
  FragU u;
#pragma unroll
  for (int i = 0; i < 16; ++i) u.e[i] = (_Float16)p[i];
  return u.v;
}

__device__ __forceinline__ float fast_tanh(float v) {
#if __has_builtin(__builtin_amdgcn_tanhf)
  return __builtin_amdgcn_tanhf(v);   // v_tanh_f32 (gfx1250 TRANS op)
#else
  return tanhf(v);
#endif
}

#define WMMA(A, B, C) \
  __builtin_amdgcn_wmma_f32_16x16x32_f16(false, (A), false, (B), (short)0, (C), false, false)

__global__ __launch_bounds__(THREADS, 1)
void VanillaRNN_42640435314711_kernel(const float* __restrict__ x,
                                      const float* __restrict__ Whx,
                                      const float* __restrict__ Whh,
                                      const float* __restrict__ Wph,
                                      const float* __restrict__ bh,
                                      const float* __restrict__ bp,
                                      float* __restrict__ out)
{
  __shared__ _Float16 whh[HID * WHH_S];     // 135168 B
  __shared__ _Float16 hs [2][BM * HS_S];    //  33792 B (double-buffered h)
  __shared__ _Float16 xs [2][BM * XS_S];    //   9216 B   total ~174 KB

  const int tid  = threadIdx.x;
  const int b0   = blockIdx.x * BM;
  const int lane = tid & 31;
  const int w    = tid >> 5;          // 16 waves
  const int l16  = lane & 15;
  const int hi   = lane >> 4;
  const int m0   = (w & 1) * 16;      // row-tile base (batch rows within block)
  const int n0   = (w >> 1) * 32;     // 2 col-tiles per wave: n0, n0+16

  // ---- stage Whh: f32 global -> f16 LDS (row-major, padded) ----
  {
    const float4* W4 = (const float4*)Whh;
#pragma unroll 1
    for (int i = tid; i < HID * HID / 4; i += THREADS) {
      float4 v = W4[i];
      int base = i * 4;
      int n = base >> 8;            // HID == 256
      int k = base & (HID - 1);
      half4 h;
      h.x = (_Float16)v.x; h.y = (_Float16)v.y;
      h.z = (_Float16)v.z; h.w = (_Float16)v.w;
      *(half4*)(&whh[n * WHH_S + k]) = h;
    }
  }
  // ---- h0 = 0 (buffer 0) ----
  for (int i = tid; i < BM * HS_S; i += THREADS) hs[0][i] = (_Float16)0.f;

  // ---- Whx B-fragments resident in registers (per wave: its 32 output cols) ----
  half16 bx[2][2];
#pragma unroll
  for (int kt = 0; kt < 2; ++kt)
#pragma unroll
    for (int j = 0; j < 2; ++j) {
      int n = n0 + j * 16 + l16;
      bx[kt][j] = frag_gl_b_f32(Whx + n * DIM + kt * 32 + hi * 16);
    }

  // ---- bias per accumulator column (C/D layout: N = l16 + 16*tile) ----
  float bias[2];
#pragma unroll
  for (int j = 0; j < 2; ++j) bias[j] = bh[n0 + j * 16 + l16];

  // ---- per-wave Whh row pointers (B operand rows) ----
  const _Float16* brow[2];
#pragma unroll
  for (int j = 0; j < 2; ++j) brow[j] = &whh[(n0 + j * 16 + l16) * WHH_S];

  // ---- x tile loader: 512 float4 per tile, 1 per thread ----
  float4 px;
  auto load_x = [&](int t) {
    int r = tid >> 4, c = (tid & 15) << 2;
    px = *(const float4*)(x + (size_t)(b0 + r) * (SEQ * DIM) + t * DIM + c);
  };
  auto store_x = [&](int buf) {
    int r = tid >> 4, c = (tid & 15) << 2;
    half4 h;
    h.x = (_Float16)px.x; h.y = (_Float16)px.y;
    h.z = (_Float16)px.z; h.w = (_Float16)px.w;
    *(half4*)(&xs[buf][r * XS_S + c]) = h;
  };

  load_x(0);
  store_x(0);
  __syncthreads();

  // ================= recurrence: 127 steps, one barrier per step =================
#pragma unroll 1
  for (int t = 0; t < SEQ - 1; ++t) {
    const int cur = t & 1;
    if (t + 1 < SEQ - 1) load_x(t + 1);   // global prefetch overlaps WMMA below

    v8f acc[2];
#pragma unroll
    for (int j = 0; j < 2; ++j) {
#pragma unroll
      for (int r = 0; r < 8; ++r) acc[j][r] = bias[j];
    }

    const _Float16* arow_h = &hs[cur][(m0 + l16) * HS_S];
    const _Float16* arow_x = &xs[cur][(m0 + l16) * XS_S];

    // ---- software-pipelined K loop: fragments for kt+1 in flight during kt ----
    half16 a0 = frag_lds_a(arow_h, 0, hi);
    half16 bb[2];
#pragma unroll
    for (int j = 0; j < 2; ++j) bb[j] = frag_lds_b(brow[j], 0, hi);

#pragma unroll
    for (int kt = 0; kt < 8; ++kt) {       // h-part: K = 256
      half16 a_n = a0;
      half16 bn[2] = {bb[0], bb[1]};
      if (kt < 7) {
        const int kb = (kt + 1) * 32;
        a_n = frag_lds_a(arow_h, kb, hi);
#pragma unroll
        for (int j = 0; j < 2; ++j) bn[j] = frag_lds_b(brow[j], kb, hi);
      } else {
        a_n = frag_lds_a(arow_x, 0, hi);   // first x fragment
      }
#pragma unroll
      for (int j = 0; j < 2; ++j) acc[j] = WMMA(a0, bb[j], acc[j]);
      a0 = a_n;
#pragma unroll
      for (int j = 0; j < 2; ++j) bb[j] = bn[j];
    }
    // x-part: K = 64, B fragments live in registers
    {
      half16 a1 = frag_lds_a(arow_x, 32, hi);
#pragma unroll
      for (int j = 0; j < 2; ++j) acc[j] = WMMA(a0, bx[0][j], acc[j]);
#pragma unroll
      for (int j = 0; j < 2; ++j) acc[j] = WMMA(a1, bx[1][j], acc[j]);
    }

    // h_new = tanh(acc) -> alternate buffer (no read/write conflict, no barrier)
    // C/D layout: M = r + 8*hi, N = l16 + 16j
#pragma unroll
    for (int j = 0; j < 2; ++j)
#pragma unroll
      for (int r = 0; r < 8; ++r)
        hs[cur ^ 1][(m0 + hi * 8 + r) * HS_S + n0 + j * 16 + l16] =
            (_Float16)fast_tanh(acc[j][r]);

    if (t + 1 < SEQ - 1) store_x(cur ^ 1);
    __syncthreads();   // h_{t+1} and x_{t+1} visible for next step
  }

  // ================= final projection: p = h @ Wph^T + bp =================
  const _Float16* hfin = hs[(SEQ - 1) & 1];
  for (int i = tid; i < BM * NCLS; i += THREADS) {
    int r = i / NCLS, c = i - r * NCLS;
    float s = bp[c];
    const _Float16* hp = &hfin[r * HS_S];
    const float* wp = Wph + c * HID;
#pragma unroll 8
    for (int k = 0; k < HID; ++k) s += (float)hp[k] * wp[k];
    out[(size_t)(b0 + r) * NCLS + c] = s;
  }
}

extern "C" void kernel_launch(void* const* d_in, const int* in_sizes, int n_in,
                              void* d_out, int out_size, void* d_ws, size_t ws_size,
                              hipStream_t stream) {
  const float* x   = (const float*)d_in[0];
  const float* Whx = (const float*)d_in[1];
  const float* Whh = (const float*)d_in[2];
  const float* Wph = (const float*)d_in[3];
  const float* bh  = (const float*)d_in[4];
  const float* bp  = (const float*)d_in[5];
  float* out = (float*)d_out;

  dim3 grid(BATCH / BM);   // 128 blocks, each owns 32 batch rows for all 127 steps
  VanillaRNN_42640435314711_kernel<<<grid, THREADS, 0, stream>>>(
      x, Whx, Whh, Wph, bh, bp, out);
}
